// VarianceAdaptor_8031588843974
// MI455X (gfx1250) — matrix operand order repaired and checked by reading, not compiled, not run on hardware
//
#include <hip/hip_runtime.h>
#include <hip/hip_bf16.h>
#include <math.h>

typedef _Float16 h16;
typedef __attribute__((ext_vector_type(16))) _Float16 v16h;
typedef __attribute__((ext_vector_type(8)))  _Float16 v8h;
typedef __attribute__((ext_vector_type(8)))  float    v8f;
typedef __attribute__((ext_vector_type(4)))  unsigned int u32x4;
typedef __attribute__((ext_vector_type(8)))  int i32x8;
typedef __attribute__((ext_vector_type(4)))  int i32x4;

#if defined(__has_builtin)
#if __has_builtin(__builtin_amdgcn_tensor_load_to_lds)
#define HAVE_TDM 1
#endif
#endif

// ---------------- problem constants ----------------
#define BN     16
#define TTEXT  512
#define HIDD   512
#define MAXLEN 2048
#define DPCH   256
#define DPK    3
#define PPCH   384
#define PPK    5
#define NBINS  256

// ---------------- workspace layout (element offsets in halfs) ----------------
#define N_WDP0 (256*512*3)
#define N_WDP1 (256*256*3)
#define N_WPP0 (384*512*5)
#define N_WPPR (4*384*384*5)
#define N_XT   (16*512*512)
#define N_PT   ((size_t)16*512*2048)
#define N_ACT  ((size_t)16*384*2048)

#define O_WDP0 ((size_t)0)
#define O_WDP1 (O_WDP0 + N_WDP0)
#define O_WPP0 (O_WDP1 + N_WDP1)
#define O_WPPR (O_WPP0 + N_WPP0)
#define O_XT   (O_WPPR + N_WPPR)
#define O_PT   (O_XT   + N_XT)
#define O_ACTA (O_PT   + N_PT)
#define O_ACTB (O_ACTA + N_ACT)
#define O_END_HALFS (O_ACTB + N_ACT)
#define O_CUM_BYTES (O_END_HALFS * 2)   // int cum[16*512] lives after the half region

// ---------------- elementwise f32 -> f16 ----------------
__global__ __launch_bounds__(256) void k_cvt(const float* __restrict__ in,
                                             h16* __restrict__ out, int n) {
    int i = blockIdx.x * 256 + threadIdx.x;
    if (i < n) out[i] = (h16)in[i];
}

// x[B,T,H] f32 -> xT[B,H,T] f16
__global__ __launch_bounds__(256) void k_transpose_cvt(const float* __restrict__ x,
                                                       h16* __restrict__ xt,
                                                       int Bn, int T, int H) {
    int i = blockIdx.x * 256 + threadIdx.x;
    int total = Bn * T * H;
    if (i >= total) return;
    int h = i % H;
    int t = (i / H) % T;
    int b = i / (H * T);
    xt[((size_t)b * H + h) * T + t] = (h16)x[i];
}

// ---------------- WMMA conv1d-as-GEMM ----------------
// Y[b, co, t] = relu( bias[co] + sum_{ci,k} W[co, ci*K+k] * X[b, ci, t+k-pad] )
// W: [Cout, KC] f16 ; X: [B, Cin, T] f16 ; Y: [B, Cout, T] f16
// block = 256 threads (8 waves), computes a 128(M) x 64(N=t) tile.
// Wave w owns rows [16w,16w+16): 1 A-frag reused over 4 WMMAs per k-step.
// A tile staged via TDM tensor_load_to_lds (TENSORcnt) when available,
// else CDNA5 async-to-LDS (ASYNCcnt). B tile is an im2col scalar gather.
__global__ __launch_bounds__(256) void k_conv_wmma(
    const h16* __restrict__ W, const h16* __restrict__ X,
    const float* __restrict__ bias, h16* __restrict__ Y,
    int Cin, int K, int KC, int Cout, int T, int pad)
{
    __shared__ __align__(16) h16 As[128 * 32];  // [m][k]
    __shared__ __align__(16) h16 Bs[64 * 32];   // [t][k]  (t-major for contiguous frags)

    const int b   = blockIdx.z;
    const int n0  = blockIdx.x * 64;      // t tile base
    const int m0  = blockIdx.y * 128;     // cout tile base
    const int tid = threadIdx.x;
    const int lane = tid & 31;
    const int wave = tid >> 5;            // 0..7 -> 16-row strip

    const h16* Xb = X + (size_t)b * Cin * T;

    v8f acc0 = {}; v8f acc1 = {}; v8f acc2 = {}; v8f acc3 = {};

    const int b_row = tid >> 2;           // 0..63
    const int b_col = (tid & 3) * 8;      // 0,8,16,24
    const int r   = lane & 15;
    const int hlf = lane >> 4;

    for (int kc0 = 0; kc0 < KC; kc0 += 32) {
#ifdef HAVE_TDM
        // ---- stage A tile (128 rows x 32 halfs) with one TDM descriptor, wave 0 only ----
        if (wave == 0) {
            unsigned long long ga =
                (unsigned long long)(uintptr_t)&W[(size_t)m0 * KC + kc0];
            u32x4 g0;
            g0[0] = 1u;                                   // count=1, user D#
            g0[1] = (unsigned)(uintptr_t)&As[0];          // lds_addr (bytes)
            g0[2] = (unsigned)ga;                         // global_addr[31:0]
            g0[3] = (unsigned)((ga >> 32) & 0x1FFFFFFull) // global_addr[56:32]
                  | (2u << 30);                           // type=2 (image)
            i32x8 g1;
            g1[0] = (int)(1u << 16);                      // data_size=2B; mask=0; no flags
            g1[1] = (int)(((unsigned)KC & 0xFFFFu) << 16);            // tensor_dim0 lo16
            g1[2] = (int)((((unsigned)KC >> 16) & 0xFFFFu)
                  | (((unsigned)Cout & 0xFFFFu) << 16));              // dim0 hi | dim1 lo
            g1[3] = (int)((((unsigned)Cout >> 16) & 0xFFFFu)
                  | (32u << 16));                                     // dim1 hi | tile_dim0=32
            g1[4] = (int)128u;                            // tile_dim1=128, tile_dim2=0
            g1[5] = (int)(unsigned)KC;                    // tensor_dim0_stride[31:0]
            g1[6] = 0;                                    // stride0 hi | stride1 lo
            g1[7] = 0;                                    // stride1 hi
            i32x4 z4 = {0, 0, 0, 0};                      // groups 2/3: zero (2D tile)
            i32x8 z8 = {0, 0, 0, 0, 0, 0, 0, 0};
            __builtin_amdgcn_tensor_load_to_lds(g0, g1, z4, z4, z8, 0);
        }
#else
        // ---- stage A tile via async-to-LDS, 2 x 16B per thread ----
        #pragma unroll
        for (int cc = 0; cc < 2; ++cc) {
            int c   = tid + cc * 256;          // chunk 0..511
            int row = c >> 2;
            int kcl = (c & 3) * 8;
            unsigned long long ga =
                (unsigned long long)(uintptr_t)&W[(size_t)(m0 + row) * KC + kc0 + kcl];
            unsigned lds = (unsigned)(uintptr_t)&As[row * 32 + kcl];
            asm volatile("global_load_async_to_lds_b128 %0, %1, off"
                         :: "v"(lds), "v"(ga) : "memory");
        }
#endif
        if (kc0 + 32 < KC)
            __builtin_prefetch(&W[(size_t)(m0 + b_row) * KC + kc0 + 32 + b_col], 0, 0);

        // ---- stage B tile: im2col gather, 8 scalar halfs per thread ----
        {
            const int tcol = n0 + b_row;
            int kg0 = kc0 + b_col;
            int ci  = kg0 / K;
            int kk  = kg0 - ci * K;
            #pragma unroll
            for (int j = 0; j < 8; ++j) {
                int ts = tcol + kk - pad;
                h16 v = (h16)0.0f;
                if (ts >= 0 && ts < T) v = Xb[(size_t)ci * T + ts];
                Bs[b_row * 32 + b_col + j] = v;
                if (++kk == K) { kk = 0; ++ci; }
            }
        }
#ifdef HAVE_TDM
        if (wave == 0) __builtin_amdgcn_s_wait_tensorcnt(0);
#else
        asm volatile("s_wait_asynccnt 0x0" ::: "memory");
#endif
        __syncthreads();

        // A fragment: lane r holds row (16*wave + r); K 0..7/16..23 (hlf=0) or 8..15/24..31 (hlf=1)
        v8h alo = *(const v8h*)&As[(16 * wave + r) * 32 + hlf * 8];
        v8h ahi = *(const v8h*)&As[(16 * wave + r) * 32 + 16 + hlf * 8];
        v16h af = __builtin_shufflevector(alo, ahi,
                   0,1,2,3,4,5,6,7,8,9,10,11,12,13,14,15);

        // 4 B fragments: lane r = column t within subtile; lane-half selects K 0..7/16..23 / 8..15/24..31
        #define BFRAG(S, DST)                                                     \
            {   v8h blo_ = *(const v8h*)&Bs[((S)*16 + r) * 32 + 8 * hlf];         \
                v8h bhi_ = *(const v8h*)&Bs[((S)*16 + r) * 32 + 16 + 8 * hlf];    \
                DST = __builtin_shufflevector(blo_, bhi_,                         \
                        0,1,2,3,4,5,6,7,8,9,10,11,12,13,14,15); }
        v16h bf0, bf1, bf2, bf3;
        BFRAG(0, bf0) BFRAG(1, bf1) BFRAG(2, bf2) BFRAG(3, bf3)
        #undef BFRAG

        acc0 = __builtin_amdgcn_wmma_f32_16x16x32_f16(false, af, false, bf0,
                                                      (short)0, acc0, false, false);
        acc1 = __builtin_amdgcn_wmma_f32_16x16x32_f16(false, af, false, bf1,
                                                      (short)0, acc1, false, false);
        acc2 = __builtin_amdgcn_wmma_f32_16x16x32_f16(false, af, false, bf2,
                                                      (short)0, acc2, false, false);
        acc3 = __builtin_amdgcn_wmma_f32_16x16x32_f16(false, af, false, bf3,
                                                      (short)0, acc3, false, false);
        __syncthreads();
    }

    // epilogue: D layout -> VGPR q, lane: M = 16*wave + q + 8*hlf, N = 16*s + r
    const int m  = m0 + 16 * wave + 8 * hlf;
    #pragma unroll
    for (int q = 0; q < 8; ++q) {
        float bv = bias[m + q];
        h16* yr = Y + ((size_t)b * Cout + (m + q)) * T + n0 + r;
        float v0 = acc0[q] + bv; v0 = v0 > 0.f ? v0 : 0.f;
        float v1 = acc1[q] + bv; v1 = v1 > 0.f ? v1 : 0.f;
        float v2 = acc2[q] + bv; v2 = v2 > 0.f ? v2 : 0.f;
        float v3 = acc3[q] + bv; v3 = v3 > 0.f ? v3 : 0.f;
        yr[0]  = (h16)v0;
        yr[16] = (h16)v1;
        yr[32] = (h16)v2;
        yr[48] = (h16)v3;
    }
}

// ---------------- channel LayerNorm (in-place, f16), optional src mask ----------------
__global__ __launch_bounds__(128) void k_ln(h16* __restrict__ buf,
                                            const float* __restrict__ g,
                                            const float* __restrict__ beta,
                                            const unsigned char* __restrict__ mask,
                                            int C, int T)
{
    int blk = blockIdx.x;
    int b = blk / T, t = blk - b * T;
    h16* col = buf + (size_t)b * C * T + t;
    __shared__ float s1[128], s2[128];
    float sum = 0.f, ss = 0.f;
    for (int c = threadIdx.x; c < C; c += 128) {
        float v = (float)col[(size_t)c * T];
        sum += v; ss += v * v;
    }
    s1[threadIdx.x] = sum; s2[threadIdx.x] = ss;
    __syncthreads();
    for (int o = 64; o > 0; o >>= 1) {
        if (threadIdx.x < o) { s1[threadIdx.x] += s1[threadIdx.x + o];
                               s2[threadIdx.x] += s2[threadIdx.x + o]; }
        __syncthreads();
    }
    float mean = s1[0] / C;
    float var  = s2[0] / C - mean * mean;
    float rstd = rsqrtf(var + 1e-5f);
    float keep = 1.f;
    if (mask) keep = mask[b * T + t] ? 0.f : 1.f;
    for (int c = threadIdx.x; c < C; c += 128) {
        float v = (float)col[(size_t)c * T];
        col[(size_t)c * T] = (h16)(((v - mean) * rstd * g[c] + beta[c]) * keep);
    }
}

// ---------------- duration linear head ----------------
__global__ __launch_bounds__(128) void k_dp_linear(const h16* __restrict__ buf,
                                                   const float* __restrict__ w,
                                                   const float* __restrict__ lb,
                                                   const unsigned char* __restrict__ mask,
                                                   float* __restrict__ out,
                                                   int C, int T)
{
    int blk = blockIdx.x;
    int b = blk / T, t = blk - b * T;
    const h16* col = buf + (size_t)b * C * T + t;
    __shared__ float s1[128];
    float sum = 0.f;
    for (int c = threadIdx.x; c < C; c += 128)
        sum += (float)col[(size_t)c * T] * w[c];
    s1[threadIdx.x] = sum;
    __syncthreads();
    for (int o = 64; o > 0; o >>= 1) {
        if (threadIdx.x < o) s1[threadIdx.x] += s1[threadIdx.x + o];
        __syncthreads();
    }
    if (threadIdx.x == 0) {
        float keep = mask[b * T + t] ? 0.f : 1.f;
        out[b * T + t] = (s1[0] + lb[0]) * keep;
    }
}

// ---------------- duration cumsum + mel_len ----------------
__global__ void k_cumsum(const int* __restrict__ dur, int* __restrict__ cum,
                         float* __restrict__ mlen_f, int Bn, int T)
{
    int b = threadIdx.x;
    if (b >= Bn) return;
    int s = 0;
    for (int t = 0; t < T; ++t) { s += dur[b * T + t]; cum[b * T + t] = s; }
    mlen_f[b] = (float)s;
}

// ---------------- length regulator: tok searchsorted + mel + pos-enc into pT ----------------
__global__ __launch_bounds__(256) void k_tok_mel(const float* __restrict__ x,
                                                 const int* __restrict__ cum,
                                                 const float* __restrict__ pos_alpha,
                                                 float* __restrict__ outmel,
                                                 h16* __restrict__ pT,
                                                 int Bn, int T, int H, int F)
{
    int blk = blockIdx.x;
    int b = blk / F, f = blk - b * F;
    const int* cb = cum + b * T;
    // searchsorted(cum, f, side='right') : first idx with cum[idx] > f
    int lo = 0, hi = T;
    while (lo < hi) { int mid = (lo + hi) >> 1; if (cb[mid] <= f) lo = mid + 1; else hi = mid; }
    int tok  = lo;
    int mlen = cb[T - 1];
    float fm = (f < mlen) ? 1.f : 0.f;
    int tk = tok < T - 1 ? tok : T - 1;
    const float* xr = x + ((size_t)b * T + tk) * H;
    float alpha = pos_alpha[0];
    const int half = H / 2;
    const float cc = -logf(10000.f) / (float)(half - 1);
    float* om = outmel + ((size_t)b * F + f) * H;
    for (int h = threadIdx.x; h < H; h += 256) {
        float m = xr[h] * fm;
        om[h] = m;
        int i = (h < half) ? h : (h - half);
        float ang = (float)(f + 1) * expf(cc * (float)i);
        float pe  = (h < half) ? sinf(ang) : cosf(ang);
        pT[((size_t)b * H + h) * F + f] = (h16)(m + alpha * pe);
    }
}

// ---------------- pitch linear head + bucketize + embedding add ----------------
__global__ __launch_bounds__(256) void k_pp_lin(const h16* __restrict__ buf,
                                                const float* __restrict__ lw,  // [C,2]
                                                const float* __restrict__ lb,  // [2]
                                                const float* __restrict__ bins,// [255]
                                                const float* __restrict__ emb, // [256,H]
                                                float* __restrict__ out_pitch,
                                                float* __restrict__ outmel,
                                                int C, int F, int H, int NB)
{
    int blk = blockIdx.x;
    int b = blk / F, f = blk - b * F;
    const h16* col = buf + (size_t)b * C * F + f;
    __shared__ float s0[256], s1[256];
    float d0 = 0.f, d1 = 0.f;
    for (int c = threadIdx.x; c < C; c += 256) {
        float v = (float)col[(size_t)c * F];
        d0 += v * lw[c * 2];
        d1 += v * lw[c * 2 + 1];
    }
    s0[threadIdx.x] = d0; s1[threadIdx.x] = d1;
    __syncthreads();
    for (int o = 128; o > 0; o >>= 1) {
        if (threadIdx.x < o) { s0[threadIdx.x] += s0[threadIdx.x + o];
                               s1[threadIdx.x] += s1[threadIdx.x + o]; }
        __syncthreads();
    }
    float v0 = s0[0] + lb[0];
    float v1 = s1[0] + lb[1];
    if (threadIdx.x == 0) {
        out_pitch[((size_t)b * F + f) * 2]     = v0;
        out_pitch[((size_t)b * F + f) * 2 + 1] = v1;
    }
    // searchsorted(bins, v0, side='left') : first idx with bins[idx] >= v0
    int lo = 0, hi = NB;
    while (lo < hi) { int mid = (lo + hi) >> 1; if (bins[mid] < v0) lo = mid + 1; else hi = mid; }
    const float* er = emb + (size_t)lo * H;
    float* om = outmel + ((size_t)b * F + f) * H;
    for (int h = threadIdx.x; h < H; h += 256) om[h] += er[h];
}

// ---------------- launcher ----------------
extern "C" void kernel_launch(void* const* d_in, const int* in_sizes, int n_in,
                              void* d_out, int out_size, void* d_ws, size_t ws_size,
                              hipStream_t stream) {
    const float* x          = (const float*)d_in[0];
    const int*   duration   = (const int*)d_in[1];
    const unsigned char* src_mask = (const unsigned char*)d_in[2];
    const float* dp_w0      = (const float*)d_in[4];
    const float* dp_w1      = (const float*)d_in[5];
    const float* dp_b       = (const float*)d_in[6];
    const float* dp_g       = (const float*)d_in[7];
    const float* dp_beta    = (const float*)d_in[8];
    const float* dp_lin_w   = (const float*)d_in[9];
    const float* dp_lin_b   = (const float*)d_in[10];
    const float* pp_w0      = (const float*)d_in[11];
    const float* pp_w_rest  = (const float*)d_in[12];
    const float* pp_b       = (const float*)d_in[13];
    const float* pp_g       = (const float*)d_in[14];
    const float* pp_beta    = (const float*)d_in[15];
    const float* pp_lin_w   = (const float*)d_in[16];
    const float* pp_lin_b   = (const float*)d_in[17];
    const float* pos_alpha  = (const float*)d_in[18];
    const float* pitch_emb  = (const float*)d_in[19];
    const float* pitch_bins = (const float*)d_in[20];

    h16* hws = (h16*)d_ws;
    h16* wdp0 = hws + O_WDP0;
    h16* wdp1 = hws + O_WDP1;
    h16* wpp0 = hws + O_WPP0;
    h16* wppr = hws + O_WPPR;
    h16* xT   = hws + O_XT;
    h16* pT   = hws + O_PT;
    h16* actA = hws + O_ACTA;
    h16* actB = hws + O_ACTB;
    int* cum  = (int*)((char*)d_ws + O_CUM_BYTES);

    float* out        = (float*)d_out;
    float* out_logdur = out + (size_t)BN * MAXLEN * HIDD;
    float* out_pitch  = out_logdur + BN * TTEXT;
    float* out_mellen = out_pitch + (size_t)BN * MAXLEN * 2;

    // weight + input conversions to f16
    k_cvt<<<(N_WDP0 + 255) / 256, 256, 0, stream>>>(dp_w0, wdp0, N_WDP0);
    k_cvt<<<(N_WDP1 + 255) / 256, 256, 0, stream>>>(dp_w1, wdp1, N_WDP1);
    k_cvt<<<(N_WPP0 + 255) / 256, 256, 0, stream>>>(pp_w0, wpp0, N_WPP0);
    k_cvt<<<(N_WPPR + 255) / 256, 256, 0, stream>>>(pp_w_rest, wppr, N_WPPR);
    k_transpose_cvt<<<(N_XT + 255) / 256, 256, 0, stream>>>(x, xT, BN, TTEXT, HIDD);

    // -------- duration predictor --------
    k_conv_wmma<<<dim3(TTEXT / 64, DPCH / 128, BN), 256, 0, stream>>>(
        wdp0, xT, dp_b, actA, HIDD, DPK, HIDD * DPK, DPCH, TTEXT, (DPK - 1) / 2);
    k_ln<<<BN * TTEXT, 128, 0, stream>>>(actA, dp_g, dp_beta, src_mask, DPCH, TTEXT);
    k_conv_wmma<<<dim3(TTEXT / 64, DPCH / 128, BN), 256, 0, stream>>>(
        wdp1, actA, dp_b + DPCH, actB, DPCH, DPK, DPCH * DPK, DPCH, TTEXT, (DPK - 1) / 2);
    k_ln<<<BN * TTEXT, 128, 0, stream>>>(actB, dp_g + DPCH, dp_beta + DPCH, src_mask, DPCH, TTEXT);
    k_dp_linear<<<BN * TTEXT, 128, 0, stream>>>(actB, dp_lin_w, dp_lin_b, src_mask,
                                                out_logdur, DPCH, TTEXT);

    // -------- length regulator --------
    k_cumsum<<<1, 32, 0, stream>>>(duration, cum, out_mellen, BN, TTEXT);
    k_tok_mel<<<BN * MAXLEN, 256, 0, stream>>>(x, cum, pos_alpha, out, pT,
                                               BN, TTEXT, HIDD, MAXLEN);

    // -------- pitch predictor --------
    k_conv_wmma<<<dim3(MAXLEN / 64, PPCH / 128, BN), 256, 0, stream>>>(
        wpp0, pT, pp_b, actA, HIDD, PPK, HIDD * PPK, PPCH, MAXLEN, (PPK - 1) / 2);
    k_ln<<<BN * MAXLEN, 128, 0, stream>>>(actA, pp_g, pp_beta, nullptr, PPCH, MAXLEN);
    for (int i = 1; i < 5; ++i) {
        h16* src = (i & 1) ? actA : actB;
        h16* dst = (i & 1) ? actB : actA;
        k_conv_wmma<<<dim3(MAXLEN / 64, PPCH / 128, BN), 256, 0, stream>>>(
            wppr + (size_t)(i - 1) * PPCH * PPCH * PPK, src, pp_b + i * PPCH, dst,
            PPCH, PPK, PPCH * PPK, PPCH, MAXLEN, (PPK - 1) / 2);
        k_ln<<<BN * MAXLEN, 128, 0, stream>>>(dst, pp_g + i * PPCH, pp_beta + i * PPCH,
                                              nullptr, PPCH, MAXLEN);
    }
    // final activation is in actA (i=4 wrote dst=actA)
    k_pp_lin<<<BN * MAXLEN, 256, 0, stream>>>(actA, pp_lin_w, pp_lin_b, pitch_bins,
                                              pitch_emb, out_pitch, out,
                                              PPCH, MAXLEN, HIDD, NBINS - 1);
}
